// BlockLTP_13554916786622
// MI455X (gfx1250) — matrix operand order repaired and checked by reading, not compiled, not run on hardware
//
#include <hip/hip_runtime.h>
#include <hip/hip_bf16.h>

// ---------------------------------------------------------------------------
// CDNA5 (gfx1250) WMMA types
// ---------------------------------------------------------------------------
typedef __attribute__((ext_vector_type(16))) __bf16 v16bf;
typedef __attribute__((ext_vector_type(8))) float v8f;

union FragBF {
  v16bf v;
  uint4 u[2];
};

__device__ __forceinline__ v8f wmma_bf16(v16bf a, v16bf b, v8f c) {
  // D = A(16x32 bf16) x B(32x16 bf16) + C(16x16 f32)
  return __builtin_amdgcn_wmma_f32_16x16x32_bf16(false, a, false, b, (short)0, c,
                                                 false, false);
}

// ---------------------------------------------------------------------------
// CDNA5 async global->LDS copy (ASYNCcnt path, CDNA5 ISA)
// GV mode: 64-bit VGPR address, saddr=off. Per-lane 16B copy into LDS.
// ---------------------------------------------------------------------------
__device__ __forceinline__ void async_ld_b128(void* lds_dst, const void* gaddr) {
  const unsigned ldsa = (unsigned)(size_t)lds_dst;  // low 32 bits = LDS offset
  asm volatile("global_load_async_to_lds_b128 %0, %1, off"
               :
               : "v"(ldsa), "v"(gaddr)
               : "memory");
}
__device__ __forceinline__ void wait_asynccnt_0() {
  asm volatile("s_wait_asynccnt 0x0" ::: "memory");
}
__device__ __forceinline__ void wait_asynccnt_4() {
  asm volatile("s_wait_asynccnt 0x4" ::: "memory");
}

constexpr int kB = 4, kT = 1024, kC = 1024, kH = 16, kD = 64;
constexpr int kM = kB * kT;  // 4096 token rows
constexpr int k3C = 3 * kC, k4C = 4 * kC;

// ---------------------------------------------------------------------------
// f32 [K][N] -> bf16 transposed [N][K] (weights; makes GEMM B-tiles memcpy-able)
// ---------------------------------------------------------------------------
__global__ void transpose_to_bf16_kernel(const float* __restrict__ in,
                                         __hip_bfloat16* __restrict__ out,
                                         int K, int N) {
  int i = blockIdx.x * blockDim.x + threadIdx.x;
  const int total = K * N;
  const int stride = gridDim.x * blockDim.x;
  for (; i < total; i += stride) {
    const int n = i / K;
    const int k = i - n * K;
    out[i] = __float2bfloat16(in[(size_t)k * N + n]);
  }
}

// ---------------------------------------------------------------------------
// LayerNorm over C=1024, one row per block (128 threads), bf16 output
// ---------------------------------------------------------------------------
__global__ __launch_bounds__(128) void ln_kernel(const float* __restrict__ x,
                                                 const float* __restrict__ w,
                                                 const float* __restrict__ bsh,
                                                 __hip_bfloat16* __restrict__ out) {
  __shared__ float r1[128];
  __shared__ float r2[128];
  const int row = blockIdx.x;
  const int t = threadIdx.x;
  const float* xr = x + (size_t)row * kC;
  float v[8];
  float s = 0.f, s2 = 0.f;
#pragma unroll
  for (int j = 0; j < 8; ++j) {
    v[j] = xr[t + j * 128];
    s += v[j];
    s2 += v[j] * v[j];
  }
  r1[t] = s;
  r2[t] = s2;
  __syncthreads();
  for (int off = 64; off > 0; off >>= 1) {
    if (t < off) {
      r1[t] += r1[t + off];
      r2[t] += r2[t + off];
    }
    __syncthreads();
  }
  const float mu = r1[0] * (1.0f / kC);
  const float var = r2[0] * (1.0f / kC) - mu * mu;
  const float rstd = rsqrtf(var + 1e-5f);
#pragma unroll
  for (int j = 0; j < 8; ++j) {
    const int c = t + j * 128;
    out[(size_t)row * kC + c] = __float2bfloat16((v[j] - mu) * rstd * w[c] + bsh[c]);
  }
}

// ---------------------------------------------------------------------------
// Tiled BF16 WMMA GEMM: C[M,N] = A[M,K] * Bt[N,K]^T (+bias, fused epilogues)
// Block tile 128x128, 8 waves (2x4), each wave 64x32 = 4x2 16x16 WMMA tiles.
// Double-buffered LDS tiles staged with global_load_async_to_lds_b128.
// EPI 0: +bias -> bf16                 (QKV)
// EPI 1: +bias +res, *rowscale -> f32  (attn proj + residual + prune mask)
// EPI 2: +bias, exact GELU -> bf16     (fc)
// EPI 3: +bias +res -> f32             (fc2 -> final output)
// ---------------------------------------------------------------------------
template <int EPI>
__global__ __launch_bounds__(256) void gemm_bf16_kernel(
    const __hip_bfloat16* __restrict__ A, const __hip_bfloat16* __restrict__ Bt,
    const float* __restrict__ bias, const float* __restrict__ res,
    const float* __restrict__ rowscale, float* __restrict__ outF,
    __hip_bfloat16* __restrict__ outB, int M, int N, int K) {
  __shared__ __align__(16) __hip_bfloat16 sA[2][128 * 32];  // [m][k]
  __shared__ __align__(16) __hip_bfloat16 sB[2][128 * 32];  // [n][k]

  const int tid = threadIdx.x;
  const int lane = tid & 31;
  const int wid = tid >> 5;
  const int waveM = wid >> 2;  // 0..1
  const int waveN = wid & 3;   // 0..3
  const int half = lane >> 4;  // lane half (wave32)
  const int l16 = lane & 15;
  const int m0 = blockIdx.y * 128;
  const int n0 = blockIdx.x * 128;

  // per-thread staging chunk coords: 512 16B chunks per tile, 2 per thread
  const int c0 = tid;
  const int c1 = tid + 256;
  const int am0 = c0 >> 2, as0 = (c0 & 3) * 8;
  const int am1 = c1 >> 2, as1 = (c1 & 3) * 8;

  v8f acc[4][2] = {};

  // prologue: async-stage k0=0 into buffer 0
  {
    uint4* dA = (uint4*)sA[0];
    uint4* dB = (uint4*)sB[0];
    async_ld_b128(dA + c0, A + (size_t)(m0 + am0) * K + as0);
    async_ld_b128(dA + c1, A + (size_t)(m0 + am1) * K + as1);
    async_ld_b128(dB + c0, Bt + (size_t)(n0 + am0) * K + as0);
    async_ld_b128(dB + c1, Bt + (size_t)(n0 + am1) * K + as1);
  }

  int cur = 0;
  for (int k0 = 0; k0 < K; k0 += 32) {
    if (k0 + 32 < K) {
      // prefetch next tile into the other buffer, then wait only for current
      uint4* dA = (uint4*)sA[cur ^ 1];
      uint4* dB = (uint4*)sB[cur ^ 1];
      const int kn = k0 + 32;
      async_ld_b128(dA + c0, A + (size_t)(m0 + am0) * K + kn + as0);
      async_ld_b128(dA + c1, A + (size_t)(m0 + am1) * K + kn + as1);
      async_ld_b128(dB + c0, Bt + (size_t)(n0 + am0) * K + kn + as0);
      async_ld_b128(dB + c1, Bt + (size_t)(n0 + am1) * K + kn + as1);
      wait_asynccnt_4();  // loads complete in order: current tile is resident
    } else {
      wait_asynccnt_0();
    }
    __syncthreads();

    const __hip_bfloat16* tA = sA[cur];
    const __hip_bfloat16* tB = sB[cur];
    FragBF a[4], bfr[2];
#pragma unroll
    for (int ms = 0; ms < 4; ++ms) {
      // A frag: lane half holds K {0-7,16-23} / {8-15,24-31}
      const __hip_bfloat16* p = tA + (waveM * 64 + ms * 16 + l16) * 32 + half * 8;
      a[ms].u[0] = *(const uint4*)p;
      a[ms].u[1] = *(const uint4*)(p + 16);
    }
#pragma unroll
    for (int ns = 0; ns < 2; ++ns) {
      // B frag: lane half holds K {0-7,16-23} / {8-15,24-31}
      const __hip_bfloat16* p = tB + (waveN * 32 + ns * 16 + l16) * 32 + half * 8;
      bfr[ns].u[0] = *(const uint4*)p;
      bfr[ns].u[1] = *(const uint4*)(p + 16);
    }
#pragma unroll
    for (int ms = 0; ms < 4; ++ms)
#pragma unroll
      for (int ns = 0; ns < 2; ++ns)
        acc[ms][ns] = wmma_bf16(a[ms].v, bfr[ns].v, acc[ms][ns]);

    __syncthreads();
    cur ^= 1;
  }

  // epilogue: C layout VGPR r -> row r (lanes<16) / r+8 (lanes>=16), col = l16
#pragma unroll
  for (int ms = 0; ms < 4; ++ms) {
#pragma unroll
    for (int ns = 0; ns < 2; ++ns) {
      const int col = n0 + waveN * 32 + ns * 16 + l16;
      const float bv = bias[col];
#pragma unroll
      for (int r = 0; r < 8; ++r) {
        const int row = m0 + waveM * 64 + ms * 16 + r + half * 8;
        const size_t idx = (size_t)row * N + col;
        const float v = acc[ms][ns][r] + bv;
        if (EPI == 0) {
          outB[idx] = __float2bfloat16(v);
        } else if (EPI == 1) {
          outF[idx] = (res[idx] + v) * rowscale[row];
        } else if (EPI == 2) {
          outB[idx] = __float2bfloat16(0.5f * v * (1.0f + erff(v * 0.70710678118f)));
        } else {
          outF[idx] = res[idx] + v;
        }
      }
    }
  }
}

// ---------------------------------------------------------------------------
// Flash attention per (b,h,64-query tile). 4 waves, one 16-row tile each.
// Pass 1: online-softmax y = softmax(QK^T/sqrt(D)) V  (causal + key mask)
// Pass 2: recompute S, accumulate exact normalized column sums (importance)
// K tiles staged with async global->LDS copies; V^T staged manually.
// ---------------------------------------------------------------------------
__global__ __launch_bounds__(128) void attn_kernel(
    const __hip_bfloat16* __restrict__ qkv, const float* __restrict__ amask,
    __hip_bfloat16* __restrict__ y, float* __restrict__ imp) {
  __shared__ __align__(16) __hip_bfloat16 sK[64 * 64];      // [key][d]
  __shared__ __align__(16) __hip_bfloat16 sVt[64 * 64];     // [d][key]
  __shared__ __align__(16) __hip_bfloat16 sP[4 * 16 * 64];  // per-wave P tile
  __shared__ float sKm[64];

  const int tid = threadIdx.x;
  const int lane = tid & 31;
  const int wid = tid >> 5;
  const int half = lane >> 4;
  const int l16 = lane & 15;
  const int qt = blockIdx.x;
  const int bh = blockIdx.y;
  const int b = bh >> 4;
  const int h = bh & 15;
  const int qr0 = qt * 64 + wid * 16;
  const float scale = 0.125f;  // 1/sqrt(64)

  // Q fragments, kept in registers (A layout, 2 chunks of K=32 over D=64)
  FragBF aq[2];
  {
    const __hip_bfloat16* qp = qkv + (size_t)(b * kT + qr0 + l16) * k3C + h * kD;
#pragma unroll
    for (int c = 0; c < 2; ++c) {
      const __hip_bfloat16* p = qp + c * 32 + half * 8;
      aq[c].u[0] = *(const uint4*)p;
      aq[c].u[1] = *(const uint4*)(p + 16);
    }
  }

  float m8[8], l8[8];
  v8f o[4] = {};
#pragma unroll
  for (int r = 0; r < 8; ++r) {
    m8[r] = -1e30f;
    l8[r] = 0.f;
  }

  for (int kt = 0; kt <= qt; ++kt) {
    __syncthreads();
    // K tile: contiguous 128B rows -> async 16B chunk copies (512 chunks)
#pragma unroll
    for (int j = 0; j < 4; ++j) {
      const int c = tid + j * 128;
      const int key = c >> 3;
      const int seg = (c & 7) * 8;
      async_ld_b128((uint4*)sK + c,
                    qkv + (size_t)(b * kT + kt * 64 + key) * k3C + kC + h * kD + seg);
    }
    // V tile: transpose into LDS while staging
#pragma unroll
    for (int j = 0; j < 16; ++j) {
      const int c = tid + j * 128;  // 2048 dword pairs
      const int key = c >> 5;
      const int dp = c & 31;
      const unsigned vv = *(const unsigned*)(
          qkv + (size_t)(b * kT + kt * 64 + key) * k3C + 2 * kC + h * kD + dp * 2);
      ((unsigned short*)sVt)[(dp * 2) * 64 + key] = (unsigned short)(vv & 0xffffu);
      ((unsigned short*)sVt)[(dp * 2 + 1) * 64 + key] = (unsigned short)(vv >> 16);
    }
    if (tid < 64) sKm[tid] = amask[b * kT + kt * 64 + tid];
    wait_asynccnt_0();
    __syncthreads();

    // S = Q K^T (scaled later)
    v8f s[4] = {};
#pragma unroll
    for (int ns = 0; ns < 4; ++ns) {
#pragma unroll
      for (int c = 0; c < 2; ++c) {
        FragBF bk;
        const __hip_bfloat16* p = sK + (ns * 16 + l16) * 64 + c * 32 + half * 8;
        bk.u[0] = *(const uint4*)p;
        bk.u[1] = *(const uint4*)(p + 16);
        s[ns] = wmma_bf16(aq[c].v, bk.v, s[ns]);
      }
    }
    // scale + causal + key-validity mask
#pragma unroll
    for (int ns = 0; ns < 4; ++ns) {
      const int kg = kt * 64 + ns * 16 + l16;
      const float km = sKm[ns * 16 + l16];
#pragma unroll
      for (int r = 0; r < 8; ++r) {
        const int qg = qr0 + r + half * 8;
        float sv = s[ns][r] * scale;
        if (kg > qg || km == 0.f) sv = -1e30f;
        s[ns][r] = sv;
      }
    }
    // online softmax: rows live in half-wave lane groups (C layout)
    float corr[8];
#pragma unroll
    for (int r = 0; r < 8; ++r) {
      float v = fmaxf(fmaxf(s[0][r], s[1][r]), fmaxf(s[2][r], s[3][r]));
#pragma unroll
      for (int off = 1; off < 16; off <<= 1) v = fmaxf(v, __shfl_xor(v, off, 32));
      const float mn = fmaxf(m8[r], v);
      corr[r] = __expf(m8[r] - mn);
      float sum = 0.f;
#pragma unroll
      for (int ns = 0; ns < 4; ++ns) {
        const float p = __expf(s[ns][r] - mn);
        s[ns][r] = p;
        sum += p;
      }
#pragma unroll
      for (int off = 1; off < 16; off <<= 1) sum += __shfl_xor(sum, off, 32);
      l8[r] = l8[r] * corr[r] + sum;
      m8[r] = mn;
    }
#pragma unroll
    for (int d = 0; d < 4; ++d)
#pragma unroll
      for (int r = 0; r < 8; ++r) o[d][r] *= corr[r];

    // stage P (C layout -> A layout) through LDS, per-wave region
    __hip_bfloat16* pw = sP + wid * 1024;
#pragma unroll
    for (int ns = 0; ns < 4; ++ns)
#pragma unroll
      for (int r = 0; r < 8; ++r)
        pw[(r + half * 8) * 64 + ns * 16 + l16] = __float2bfloat16(s[ns][r]);
    __syncthreads();

    // O += P V
#pragma unroll
    for (int kc = 0; kc < 2; ++kc) {
      FragBF ap;
      const __hip_bfloat16* p = pw + l16 * 64 + kc * 32 + half * 8;
      ap.u[0] = *(const uint4*)p;
      ap.u[1] = *(const uint4*)(p + 16);
#pragma unroll
      for (int d = 0; d < 4; ++d) {
        FragBF bv;
        const __hip_bfloat16* vp = sVt + (d * 16 + l16) * 64 + kc * 32 + half * 8;
        bv.u[0] = *(const uint4*)vp;
        bv.u[1] = *(const uint4*)(vp + 16);
        o[d] = wmma_bf16(ap.v, bv.v, o[d]);
      }
    }
  }

  // normalize + write y (B,T,H*D)
  float rl8[8];
#pragma unroll
  for (int r = 0; r < 8; ++r) rl8[r] = 1.0f / l8[r];
#pragma unroll
  for (int d = 0; d < 4; ++d) {
#pragma unroll
    for (int r = 0; r < 8; ++r) {
      const int row = qr0 + r + half * 8;
      y[(size_t)(b * kT + row) * kC + h * kD + d * 16 + l16] =
          __float2bfloat16(o[d][r] * rl8[r]);
    }
  }

  // pass 2: exact importance column sums with final m,l
  for (int kt = 0; kt <= qt; ++kt) {
    __syncthreads();
#pragma unroll
    for (int j = 0; j < 4; ++j) {
      const int c = tid + j * 128;
      const int key = c >> 3;
      const int seg = (c & 7) * 8;
      async_ld_b128((uint4*)sK + c,
                    qkv + (size_t)(b * kT + kt * 64 + key) * k3C + kC + h * kD + seg);
    }
    if (tid < 64) sKm[tid] = amask[b * kT + kt * 64 + tid];
    wait_asynccnt_0();
    __syncthreads();

    v8f s[4] = {};
#pragma unroll
    for (int ns = 0; ns < 4; ++ns) {
#pragma unroll
      for (int c = 0; c < 2; ++c) {
        FragBF bk;
        const __hip_bfloat16* p = sK + (ns * 16 + l16) * 64 + c * 32 + half * 8;
        bk.u[0] = *(const uint4*)p;
        bk.u[1] = *(const uint4*)(p + 16);
        s[ns] = wmma_bf16(aq[c].v, bk.v, s[ns]);
      }
    }
#pragma unroll
    for (int ns = 0; ns < 4; ++ns) {
      const int kg = kt * 64 + ns * 16 + l16;
      const float km = sKm[ns * 16 + l16];
      float cs = 0.f;
#pragma unroll
      for (int r = 0; r < 8; ++r) {
        const int qg = qr0 + r + half * 8;
        const float sv = s[ns][r] * scale;
        const float p =
            (kg > qg || km == 0.f) ? 0.f : __expf(sv - m8[r]) * rl8[r];
        cs += p;
      }
      cs += __shfl_xor(cs, 16, 32);  // combine both row halves -> full col sum
      if (half == 0) atomicAdd(&imp[b * kT + kg], cs);
    }
  }
}

// ---------------------------------------------------------------------------
// combined pruning mask + pruning_loss output
// ---------------------------------------------------------------------------
__global__ void mask_kernel(const float* __restrict__ imp,
                            const float* __restrict__ amask,
                            const float* __restrict__ thr,
                            float* __restrict__ outMask,
                            float* __restrict__ outLoss) {
  const int i = blockIdx.x * blockDim.x + threadIdx.x;
  if (i < kM) {
    const float mean = imp[i] * (1.0f / (float)(kH * kT));
    const float pm = (mean >= thr[0]) ? 1.f : 0.f;
    outMask[i] = amask[i] * pm;
  }
  if (i == 0) outLoss[0] = 0.f;
}

// ---------------------------------------------------------------------------
extern "C" void kernel_launch(void* const* d_in, const int* in_sizes, int n_in,
                              void* d_out, int out_size, void* d_ws,
                              size_t ws_size, hipStream_t stream) {
  (void)in_sizes;
  (void)n_in;
  (void)out_size;
  (void)ws_size;

  const float* x = (const float*)d_in[0];
  const float* amask = (const float*)d_in[1];
  const float* ln1w = (const float*)d_in[2];
  const float* ln1b = (const float*)d_in[3];
  const float* w_attn = (const float*)d_in[4];
  const float* b_attn = (const float*)d_in[5];
  const float* w_proj = (const float*)d_in[6];
  const float* b_proj = (const float*)d_in[7];
  const float* thr = (const float*)d_in[8];
  const float* ln2w = (const float*)d_in[9];
  const float* ln2b = (const float*)d_in[10];
  const float* w_fc = (const float*)d_in[11];
  const float* b_fc = (const float*)d_in[12];
  const float* w_fc2 = (const float*)d_in[13];
  const float* b_fc2 = (const float*)d_in[14];

  char* ws = (char*)d_ws;
  size_t off = 0;
  auto take = [&](size_t bytes) -> void* {
    void* p = (void*)(ws + off);
    off += (bytes + 255) & ~(size_t)255;
    return p;
  };
  // transposed bf16 weights: [N][K]
  __hip_bfloat16* wattn_t = (__hip_bfloat16*)take((size_t)kC * k3C * 2);
  __hip_bfloat16* wproj_t = (__hip_bfloat16*)take((size_t)kC * kC * 2);
  __hip_bfloat16* wfc_t = (__hip_bfloat16*)take((size_t)kC * k4C * 2);
  __hip_bfloat16* wfc2_t = (__hip_bfloat16*)take((size_t)k4C * kC * 2);
  __hip_bfloat16* hb = (__hip_bfloat16*)take((size_t)kM * kC * 2);
  __hip_bfloat16* qkvb = (__hip_bfloat16*)take((size_t)kM * k3C * 2);
  __hip_bfloat16* yb = (__hip_bfloat16*)take((size_t)kM * kC * 2);
  float* imp = (float*)take((size_t)kM * 4);
  float* x1 = (float*)take((size_t)kM * kC * 4);
  __hip_bfloat16* h2b = (__hip_bfloat16*)take((size_t)kM * kC * 2);
  __hip_bfloat16* fb = (__hip_bfloat16*)take((size_t)kM * k4C * 2);

  float* outX = (float*)d_out;
  float* outMask = outX + (size_t)kM * kC;
  float* outLoss = outMask + kM;

  hipMemsetAsync(imp, 0, (size_t)kM * 4, stream);

  transpose_to_bf16_kernel<<<1024, 256, 0, stream>>>(w_attn, wattn_t, kC, k3C);
  transpose_to_bf16_kernel<<<1024, 256, 0, stream>>>(w_proj, wproj_t, kC, kC);
  transpose_to_bf16_kernel<<<1024, 256, 0, stream>>>(w_fc, wfc_t, kC, k4C);
  transpose_to_bf16_kernel<<<1024, 256, 0, stream>>>(w_fc2, wfc2_t, k4C, kC);

  ln_kernel<<<kM, 128, 0, stream>>>(x, ln1w, ln1b, hb);

  gemm_bf16_kernel<0><<<dim3(k3C / 128, kM / 128), 256, 0, stream>>>(
      hb, wattn_t, b_attn, nullptr, nullptr, nullptr, qkvb, kM, k3C, kC);

  attn_kernel<<<dim3(kT / 64, kB * kH), 128, 0, stream>>>(qkvb, amask, yb, imp);

  mask_kernel<<<16, 256, 0, stream>>>(imp, amask, thr, outMask, outLoss);

  gemm_bf16_kernel<1><<<dim3(kC / 128, kM / 128), 256, 0, stream>>>(
      yb, wproj_t, b_proj, x, outMask, x1, nullptr, kM, kC, kC);

  ln_kernel<<<kM, 128, 0, stream>>>(x1, ln2w, ln2b, h2b);

  gemm_bf16_kernel<2><<<dim3(k4C / 128, kM / 128), 256, 0, stream>>>(
      h2b, wfc_t, b_fc, nullptr, nullptr, nullptr, fb, kM, k4C, kC);

  gemm_bf16_kernel<3><<<dim3(kC / 128, kM / 128), 256, 0, stream>>>(
      fb, wfc2_t, b_fc2, x1, nullptr, outX, nullptr, kM, kC, k4C);
}
